// Attention_60928406061617
// MI455X (gfx1250) — matrix operand order repaired and checked by reading, not compiled, not run on hardware
//
#include <hip/hip_runtime.h>
#include <hip/hip_bf16.h>
#include <math.h>

typedef __bf16 bf16;
typedef __attribute__((ext_vector_type(8)))  bf16  v8bf;
typedef __attribute__((ext_vector_type(16))) bf16  v16bf;
typedef __attribute__((ext_vector_type(8)))  float v8f;

#define BB 16
#define NN 577
#define CC 768
#define HH 12
#define DD 64
#define NP 608           // padded sequence length: 38*16, 19*32
#define C3 2304
#define MM (BB*NN)       // 9232 rows = 577 tiles of 16 (exact)

static __device__ inline bf16  f2bf(float f){ return (bf16)f; }
static __device__ inline float bf2f(bf16 b){ return (float)b; }

static __device__ inline v16bf concat8(v8bf lo, v8bf hi){
  return __builtin_shufflevector(lo, hi, 0,1,2,3,4,5,6,7,8,9,10,11,12,13,14,15);
}

// A-matrix fragment (16x32 bf16, M x K), A row-major with K contiguous.
// lane L (L<16) holds row M=L, K in {kh..kh+7, 16+kh..16+kh+7}, kh = 8*(L/16).
static __device__ inline v16bf load_afrag(const bf16* A, int lda, int row0, int k0, int lane){
  int r  = row0 + (lane & 15);
  int kh = (lane >> 4) << 3;
  const bf16* p = A + (size_t)r * (size_t)lda + (size_t)(k0 + kh);
  v8bf lo = *(const v8bf*)p;         // K = k0+kh .. +7
  v8bf hi = *(const v8bf*)(p + 16);  // K = k0+16+kh .. +7
  return concat8(lo, hi);
}

// B-matrix fragment (32x16 bf16, K x N), supplied as Bt = B^T row-major
// (row = output column, K contiguous). lane L holds column L%16,
// K in {kh..kh+7, 16+kh..16+kh+7}, kh = 8*(L/16).
static __device__ inline v16bf load_bfrag(const bf16* Bt, int ldb, int col0, int k0, int lane){
  int c  = col0 + (lane & 15);
  int kr = (lane >> 4) << 3;
  const bf16* p = Bt + (size_t)c * (size_t)ldb + (size_t)(k0 + kr);
  v8bf lo = *(const v8bf*)p;
  v8bf hi = *(const v8bf*)(p + 16);
  return concat8(lo, hi);
}

static __device__ inline v8f wmma_bf(v16bf a, v16bf b, v8f c){
  return __builtin_amdgcn_wmma_f32_16x16x32_bf16(false, a, false, b, (short)0, c, false, false);
}

// ---------------- utility kernels ----------------
__global__ void cvt_f32_bf16(const float* __restrict__ in, bf16* __restrict__ out, int n){
  int i = blockIdx.x * blockDim.x + threadIdx.x;
  if (i < n) out[i] = f2bf(in[i]);
}
__global__ void zero_u32(unsigned int* __restrict__ p, int n){
  int i = blockIdx.x * blockDim.x + threadIdx.x;
  if (i < n) p[i] = 0u;
}

// ---------------- GEMM1: qkv = x @ Wqkv^T, scatter to q/k/vT ----------------
__global__ void gemm_qkv(const bf16* __restrict__ X, const bf16* __restrict__ W,
                         bf16* __restrict__ q, bf16* __restrict__ k, bf16* __restrict__ vT){
  int wave = (blockIdx.x * blockDim.x + threadIdx.x) >> 5;
  if (wave >= (MM/16) * (C3/64)) return;           // wave-uniform
  int lane = threadIdx.x & 31;
  int mt = wave / (C3/64), ns = wave % (C3/64);
  int row0 = mt * 16, col0 = ns * 64;
  v8f acc[4] = {};
  for (int k0 = 0; k0 < CC; k0 += 32){
    v16bf a = load_afrag(X, CC, row0, k0, lane);
#pragma unroll
    for (int f = 0; f < 4; f++){
      v16bf b = load_bfrag(W, CC, col0 + 16*f, k0, lane);
      acc[f] = wmma_bf(a, b, acc[f]);
    }
  }
  int t = col0 / CC;                // 0=q, 1=k, 2=v
  int h = (col0 % CC) / DD;         // 64-col strips align exactly with heads
#pragma unroll
  for (int f = 0; f < 4; f++){
#pragma unroll
    for (int r = 0; r < 8; r++){
      int row = row0 + r + ((lane >> 4) << 3);
      int dd  = 16*f + (lane & 15);
      int b_  = row / NN, n = row % NN;
      bf16 val = f2bf(acc[f][r]);
      size_t bh = (size_t)(b_ * HH + h);
      if      (t == 0) q [(bh * NP + n) * DD + dd] = val;
      else if (t == 1) k [(bh * NP + n) * DD + dd] = val;
      else             vT[(bh * DD + dd) * NP + n] = val;
    }
  }
}

// ---------------- GEMM2: S = scale * q @ k^T per (b,h) ----------------
__global__ void gemm_scores(const bf16* __restrict__ q, const bf16* __restrict__ k,
                            bf16* __restrict__ S){
  int wave = (blockIdx.x * blockDim.x + threadIdx.x) >> 5;
  const int MT = NP/16, NS = NP/32;                 // 38, 19
  if (wave >= BB*HH*MT*NS) return;
  int lane = threadIdx.x & 31;
  int bh = wave / (MT*NS);
  int rem = wave % (MT*NS);
  int mt = rem / NS, ns = rem % NS;
  const bf16* qb = q + (size_t)bh * NP * DD;
  const bf16* kb = k + (size_t)bh * NP * DD;
  v8f acc[2] = {};
  for (int k0 = 0; k0 < DD; k0 += 32){
    v16bf a = load_afrag(qb, DD, mt*16, k0, lane);
#pragma unroll
    for (int f = 0; f < 2; f++){
      v16bf b = load_bfrag(kb, DD, ns*32 + 16*f, k0, lane);
      acc[f] = wmma_bf(a, b, acc[f]);
    }
  }
  const float scale = 0.125f;                       // d^-0.5, d=64
#pragma unroll
  for (int f = 0; f < 2; f++){
#pragma unroll
    for (int r = 0; r < 8; r++){
      int row = mt*16 + r + ((lane >> 4) << 3);
      int col = ns*32 + 16*f + (lane & 15);
      S[((size_t)bh * NP + row) * NP + col] = f2bf(acc[f][r] * scale);
    }
  }
}

// ---------------- fused Wpre-mix -> softmax -> Wpost-mix (in-place on S) ----------------
__global__ void mix_softmax(bf16* __restrict__ S, const float* __restrict__ Wpre,
                            const float* __restrict__ Wpost){
  __shared__ float s[HH][NP];
  __shared__ float t[HH][NP];
  __shared__ float wpre[HH*HH], wpost[HH*HH];
  int b = blockIdx.x / NN;
  int n = blockIdx.x % NN;
  int tid = threadIdx.x;

  if (tid < HH*HH) wpre[tid] = Wpre[tid];
  else if (tid < 2*HH*HH) wpost[tid - HH*HH] = Wpost[tid - HH*HH];
  for (int idx = tid; idx < HH*NN; idx += blockDim.x){
    int h = idx / NN, m = idx % NN;
    s[h][m] = bf2f(S[(((size_t)(b*HH + h)) * NP + n) * NP + m]);
  }
  __syncthreads();
  // t[g][m] = sum_h Wpre[g,h] * s[h][m]
  for (int idx = tid; idx < HH*NN; idx += blockDim.x){
    int g = idx / NN, m = idx % NN;
    float acc = 0.f;
#pragma unroll
    for (int h = 0; h < HH; h++) acc += wpre[g*HH + h] * s[h][m];
    t[g][m] = acc;
  }
  __syncthreads();
  // per-head softmax over m<NN; warp w owns head g=w (blockDim = 12 waves)
  {
    int g = tid >> 5, lane = tid & 31;
    float mx = -1e30f;
    for (int m = lane; m < NN; m += 32) mx = fmaxf(mx, t[g][m]);
#pragma unroll
    for (int off = 16; off > 0; off >>= 1) mx = fmaxf(mx, __shfl_xor(mx, off, 32));
    float sum = 0.f;
    for (int m = lane; m < NN; m += 32){ float e = __expf(t[g][m] - mx); t[g][m] = e; sum += e; }
#pragma unroll
    for (int off = 16; off > 0; off >>= 1) sum += __shfl_xor(sum, off, 32);
    float inv = 1.f / sum;
    for (int m = lane; m < NN; m += 32) t[g][m] *= inv;
  }
  __syncthreads();
  // S[g][n][m] = sum_h Wpost[g,h] * p[h][m]; zero the padded columns
  for (int idx = tid; idx < HH*NP; idx += blockDim.x){
    int g = idx / NP, m = idx % NP;
    float val = 0.f;
    if (m < NN){
#pragma unroll
      for (int h = 0; h < HH; h++) val += wpost[g*HH + h] * t[h][m];
    }
    S[(((size_t)(b*HH + g)) * NP + n) * NP + m] = f2bf(val);
  }
}

// ---------------- GEMM3: ctx = P @ v per (b,h) ----------------
__global__ void gemm_ctx(const bf16* __restrict__ S, const bf16* __restrict__ vT,
                         bf16* __restrict__ ctx){
  int wave = (blockIdx.x * blockDim.x + threadIdx.x) >> 5;
  const int MT = NP/16;                              // 38
  if (wave >= BB*HH*MT) return;
  int lane = threadIdx.x & 31;
  int bh = wave / MT, mt = wave % MT;
  const bf16* P   = S  + (size_t)bh * NP * NP;
  const bf16* vtb = vT + (size_t)bh * DD * NP;
  v8f acc[4] = {};
  for (int k0 = 0; k0 < NP; k0 += 32){
    v16bf a = load_afrag(P, NP, mt*16, k0, lane);
#pragma unroll
    for (int f = 0; f < 4; f++){
      v16bf b = load_bfrag(vtb, NP, 16*f, k0, lane);
      acc[f] = wmma_bf(a, b, acc[f]);
    }
  }
  int b_ = bh / HH, h = bh % HH;
#pragma unroll
  for (int f = 0; f < 4; f++){
#pragma unroll
    for (int r = 0; r < 8; r++){
      int n  = mt*16 + r + ((lane >> 4) << 3);
      int dd = 16*f + (lane & 15);
      if (n < NN)
        ctx[((size_t)(b_*NN + n)) * CC + h*DD + dd] = f2bf(acc[f][r]);
    }
  }
}

// ---------------- GEMM4: out = ctx @ Wproj^T + bproj (f32 out) ----------------
__global__ void gemm_proj(const bf16* __restrict__ ctx, const bf16* __restrict__ W,
                          const float* __restrict__ bias, float* __restrict__ out){
  int wave = (blockIdx.x * blockDim.x + threadIdx.x) >> 5;
  if (wave >= (MM/16) * (CC/64)) return;
  int lane = threadIdx.x & 31;
  int mt = wave / (CC/64), ns = wave % (CC/64);
  int row0 = mt*16, col0 = ns*64;
  v8f acc[4] = {};
  for (int k0 = 0; k0 < CC; k0 += 32){
    v16bf a = load_afrag(ctx, CC, row0, k0, lane);
#pragma unroll
    for (int f = 0; f < 4; f++){
      v16bf b = load_bfrag(W, CC, col0 + 16*f, k0, lane);
      acc[f] = wmma_bf(a, b, acc[f]);
    }
  }
#pragma unroll
  for (int f = 0; f < 4; f++){
#pragma unroll
    for (int r = 0; r < 8; r++){
      int row = row0 + r + ((lane >> 4) << 3);
      int col = col0 + 16*f + (lane & 15);
      out[(size_t)row * CC + col] = acc[f][r] + bias[col];
    }
  }
}

// ---------------- host ----------------
extern "C" void kernel_launch(void* const* d_in, const int* in_sizes, int n_in,
                              void* d_out, int out_size, void* d_ws, size_t ws_size,
                              hipStream_t stream) {
  (void)in_sizes; (void)n_in; (void)out_size; (void)ws_size;
  const float* x     = (const float*)d_in[0];
  const float* Wqkv  = (const float*)d_in[1];
  const float* Wproj = (const float*)d_in[2];
  const float* bproj = (const float*)d_in[3];
  const float* Wpre  = (const float*)d_in[4];
  const float* Wpost = (const float*)d_in[5];
  float* out = (float*)d_out;
  char* ws = (char*)d_ws;

  constexpr size_t SZ_XBF   = (size_t)MM * CC * 2;          // 14,180,352
  constexpr size_t SZ_WQKV  = (size_t)C3 * CC * 2;          //  3,538,944
  constexpr size_t SZ_WPROJ = (size_t)CC * CC * 2;          //  1,179,648
  constexpr size_t SZ_QKV1  = (size_t)BB * HH * NP * DD * 2;// 14,942,208 each
  constexpr size_t SZ_S     = (size_t)BB * HH * NP * NP * 2;// 141,950,976
  constexpr size_t OFF_XBF   = 0;
  constexpr size_t OFF_WQKV  = OFF_XBF  + SZ_XBF;
  constexpr size_t OFF_WPROJ = OFF_WQKV + SZ_WQKV;
  constexpr size_t OFF_Q     = OFF_WPROJ + SZ_WPROJ;
  constexpr size_t OFF_K     = OFF_Q + SZ_QKV1;
  constexpr size_t OFF_VT    = OFF_K + SZ_QKV1;
  constexpr size_t OFF_S     = OFF_VT + SZ_QKV1;
  constexpr size_t OFF_CTX   = OFF_S + SZ_S;

  bf16* xbf  = (bf16*)(ws + OFF_XBF);
  bf16* wqkv = (bf16*)(ws + OFF_WQKV);
  bf16* wprj = (bf16*)(ws + OFF_WPROJ);
  bf16* q    = (bf16*)(ws + OFF_Q);
  bf16* k    = (bf16*)(ws + OFF_K);
  bf16* vT   = (bf16*)(ws + OFF_VT);
  bf16* S    = (bf16*)(ws + OFF_S);
  bf16* ctx  = (bf16*)(ws + OFF_CTX);

  // 1) convert inputs to bf16
  {
    int n = MM * CC;
    cvt_f32_bf16<<<(n + 255) / 256, 256, 0, stream>>>(x, xbf, n);
    n = C3 * CC;
    cvt_f32_bf16<<<(n + 255) / 256, 256, 0, stream>>>(Wqkv, wqkv, n);
    n = CC * CC;
    cvt_f32_bf16<<<(n + 255) / 256, 256, 0, stream>>>(Wproj, wprj, n);
  }
  // 2) zero q/k/vT (contiguous) so padded rows are exactly 0
  {
    int nwords = (int)(3 * SZ_QKV1 / 4);
    zero_u32<<<(nwords + 255) / 256, 256, 0, stream>>>((unsigned int*)(ws + OFF_Q), nwords);
  }
  // 3) qkv GEMM
  {
    int waves = (MM/16) * (C3/64);                  // 20772
    gemm_qkv<<<(waves + 7) / 8, 256, 0, stream>>>(xbf, wqkv, q, k, vT);
  }
  // 4) scores
  {
    int waves = BB*HH * (NP/16) * (NP/32);          // 138624
    gemm_scores<<<(waves + 7) / 8, 256, 0, stream>>>(q, k, S);
  }
  // 5) head-mix + softmax + head-mix
  mix_softmax<<<BB * NN, 384, 0, stream>>>(S, Wpre, Wpost);
  // 6) ctx = P @ v
  {
    int waves = BB*HH * (NP/16);                    // 7296
    gemm_ctx<<<(waves + 7) / 8, 256, 0, stream>>>(S, vT, ctx);
  }
  // 7) out projection
  {
    int waves = (MM/16) * (CC/64);                  // 6924
    gemm_proj<<<(waves + 7) / 8, 256, 0, stream>>>(ctx, wprj, bproj, out);
  }
}
